// DownGTOv1_22041772163218
// MI455X (gfx1250) — matrix operand order repaired and checked by reading, not compiled, not run on hardware
//
#include <hip/hip_runtime.h>
#include <hip/hip_bf16.h>

// ---------------------------------------------------------------------------
// DownGTOv1: BN+ReLU -> {3x3/s2 conv shortcut, 1x1 Q(s2)/K/V, 8-head flash
// attention (Nq=1024, Nk=4096, dk=dv=32), 1x1 Wo} -> add.
// GEMMs: v_wmma_f32_16x16x32_f16, B-tiles staged via TDM tensor_load_to_lds.
// Softmax reductions: DPP (VALU) instead of ds_bpermute.
// ---------------------------------------------------------------------------

typedef __attribute__((ext_vector_type(16))) _Float16 v16h;
typedef __attribute__((ext_vector_type(8)))  _Float16 v8h;
typedef __attribute__((ext_vector_type(8)))  float    v8f;
typedef __attribute__((ext_vector_type(4)))  unsigned u32x4;
typedef __attribute__((ext_vector_type(8)))  int      i32x8;
typedef __attribute__((ext_vector_type(4)))  int      i32x4;

#define BATCH 4
#define CIN   256
#define HH    64
#define WW    64
#define COUT  512
#define HEADS 8
#define DK    32
#define NQ    1024
#define NK    4096

// ---------------------------------------------------------------------------
// Elementwise / data-movement kernels
// ---------------------------------------------------------------------------

__global__ void bn_relu_half(const float* __restrict__ x,
                             const float* __restrict__ g,
                             const float* __restrict__ be,
                             const float* __restrict__ mu,
                             const float* __restrict__ var,
                             _Float16* __restrict__ hF) {
  int idx = blockIdx.x * blockDim.x + threadIdx.x;   // B*CIN*H*W
  int c = (idx >> 12) & 255;
  float s = g[c] * rsqrtf(var[c] + 1e-5f);
  float v = (x[idx] - mu[c]) * s + be[c];
  hF[idx] = (_Float16)fmaxf(v, 0.0f);
}

__global__ void sample_s2(const _Float16* __restrict__ hF,
                          _Float16* __restrict__ hsF) {
  int idx = blockIdx.x * blockDim.x + threadIdx.x;   // B*CIN*1024
  int sp = idx & 1023;
  int bc = idx >> 10;
  int ox = sp & 31, oy = sp >> 5;
  hsF[idx] = hF[(size_t)bc * 4096 + (size_t)(oy * 2) * 64 + ox * 2];
}

__global__ void cvt_half(const float* __restrict__ in,
                         _Float16* __restrict__ out, int n) {
  int i = blockIdx.x * blockDim.x + threadIdx.x;
  if (i < n) out[i] = (_Float16)in[i];
}

// im2col for 3x3, stride 2, pad 1: colF[b][c*9+ky*3+kx][oy*32+ox]
__global__ void im2col3x3s2(const _Float16* __restrict__ hF,
                            _Float16* __restrict__ colF) {
  int idx = blockIdx.x * blockDim.x + threadIdx.x;   // B*2304*1024
  int sp = idx & 1023;
  int rc = (idx >> 10) % 2304;
  int b  = idx / (2304 * 1024);
  int ox = sp & 31, oy = sp >> 5;
  int kx = rc % 3, ky = (rc / 3) % 3, c = rc / 9;
  int iy = oy * 2 - 1 + ky, ix = ox * 2 - 1 + kx;
  _Float16 v = (_Float16)0.0f;
  if (iy >= 0 && iy < 64 && ix >= 0 && ix < 64)
    v = hF[((size_t)b * 256 + c) * 4096 + (size_t)iy * 64 + ix];
  colF[idx] = v;
}

// in [B][256][N] (channel = h*32+kk)  -> out [B*8][N][32]
__global__ void transpose_heads(const _Float16* __restrict__ in,
                                _Float16* __restrict__ out, int N) {
  int idx = blockIdx.x * blockDim.x + threadIdx.x;   // B*256*N
  int col = idx % N;
  int c = (idx / N) & 255;
  int b = idx / (N * 256);
  int h = c >> 5, kk = c & 31;
  out[((size_t)(b * 8 + h) * N + col) * 32 + kk] = in[idx];
}

// ---------------------------------------------------------------------------
// WMMA fragment helpers (layouts per CDNA5 ISA 7.12.2)
// ---------------------------------------------------------------------------

__device__ __forceinline__ v16h pack16(v8h lo, v8h hi) {
  v16h r;
#pragma unroll
  for (int j = 0; j < 8; ++j) { r[j] = lo[j]; r[j + 8] = hi[j]; }
  return r;
}

// A fragment 16x32 f16 from row-major A[lda]: row = lane&15,
// lane<16 holds K {0..7,16..23}+k0, lane>=16 holds K {8..15,24..31}+k0.
__device__ __forceinline__ v16h load_a_frag(const _Float16* __restrict__ A,
                                            int row, int lda, int k0, int lane) {
  int khalf = (lane >> 4) << 3;
  const _Float16* p = A + (size_t)row * lda + k0 + khalf;
  return pack16(*(const v8h*)p, *(const v8h*)(p + 16));
}

// ---------------------------------------------------------------------------
// DPP 16-lane reductions (xor1, xor2 via quad_perm; xor4/8 via row mirrors)
// ---------------------------------------------------------------------------
template <int CTRL>
__device__ __forceinline__ float dpp_mov(float x) {
  int r = __builtin_amdgcn_update_dpp(__float_as_int(x), __float_as_int(x),
                                      CTRL, 0xF, 0xF, false);
  return __int_as_float(r);
}
__device__ __forceinline__ float red16_max(float x) {
  x = fmaxf(x, dpp_mov<0xB1>(x));    // quad_perm(1,0,3,2)
  x = fmaxf(x, dpp_mov<0x4E>(x));    // quad_perm(2,3,0,1)
  x = fmaxf(x, dpp_mov<0x141>(x));   // ROW_HALF_MIRROR
  x = fmaxf(x, dpp_mov<0x140>(x));   // ROW_MIRROR
  return x;
}
__device__ __forceinline__ float red16_sum(float x) {
  x += dpp_mov<0xB1>(x);
  x += dpp_mov<0x4E>(x);
  x += dpp_mov<0x141>(x);
  x += dpp_mov<0x140>(x);
  return x;
}

// ---------------------------------------------------------------------------
// TDM: 2-D tile (tileX contiguous elems x tileY rows, 2-byte data) -> LDS.
// Descriptor per CDNA5 ISA §8.3-8.5 (group0: count/lds/global/type=2,
// group1: data_size, tensor dims, tile dims, dim0 stride).
// This toolchain's builtin takes the 6-arg form (g0, g1, g2, g3, extra, cpol).
// ---------------------------------------------------------------------------
__device__ __forceinline__ void tdm_load_2d_f16(unsigned ldsOff, const void* gptr,
                                                unsigned tileX, unsigned tileY,
                                                unsigned long strideElems,
                                                unsigned tensor0, unsigned tensor1) {
  unsigned long ga = (unsigned long)gptr;
  u32x4 g0;
  g0[0] = 1u;                                         // count=1, user mode
  g0[1] = ldsOff;                                     // lds_addr (bytes)
  g0[2] = (unsigned)ga;                               // global_addr lo
  g0[3] = (unsigned)((ga >> 32) & 0x1FFFFFFu) | (2u << 30);  // hi + type=2
  i32x8 g1;
  g1[0] = (int)(1u << 16);                            // data_size = 2 bytes
  g1[1] = (int)((tensor0 & 0xFFFFu) << 16);           // tensor_dim0 lo16
  g1[2] = (int)((tensor0 >> 16) | ((tensor1 & 0xFFFFu) << 16));
  g1[3] = (int)((tensor1 >> 16) | (tileX << 16));     // tile_dim0
  g1[4] = (int)tileY;                                 // tile_dim1 (tile_dim2=0)
  g1[5] = (int)(unsigned)(strideElems & 0xFFFFFFFFu); // dim0 stride lo32
  g1[6] = (int)(unsigned)((strideElems >> 32) & 0xFFFFu);
  g1[7] = 0;
  i32x4 z4 = {0, 0, 0, 0};
  i32x8 z8 = {0, 0, 0, 0, 0, 0, 0, 0};
  __builtin_amdgcn_tensor_load_to_lds(g0, g1, z4, z4, z8, 0);
}

// ---------------------------------------------------------------------------
// WMMA GEMM: C[M,N] = A[M,K] * B[K,N] (row-major fp16 in).
// 128 threads = 4 waves; block tile 64M x 64N; wave tile 16M x 64N (4 accums).
// B chunk [32 x 64] pulled by TDM into LDS, transposed in LDS for frag reads.
// ---------------------------------------------------------------------------
template <typename OutT, bool ACCUM>
__global__ void gemm_wmma(const _Float16* __restrict__ A,
                          const _Float16* __restrict__ Bm,
                          OutT* __restrict__ C,
                          int M, int N, int K,
                          long strideB, long strideC) {
  __shared__ _Float16 ldsRaw[32][64];   // TDM destination (row-major chunk)
  __shared__ _Float16 ldsBt[64][40];    // transposed [n][kk], padded to 80B rows
  int tid = threadIdx.x;
  int lane = tid & 31;
  int wid = tid >> 5;
  int n0 = blockIdx.x * 64;
  int m0 = blockIdx.y * 64 + wid * 16;
  const _Float16* Bb = Bm + (size_t)blockIdx.z * strideB;
  OutT* Cb = C + (size_t)blockIdx.z * strideC;
  unsigned ldsRawOff = (unsigned)(size_t)&ldsRaw[0][0];

  v8f acc[4] = {};
  for (int k0 = 0; k0 < K; k0 += 32) {
    __syncthreads();                       // previous iteration's readers done
    if (wid == 0) {
      tdm_load_2d_f16(ldsRawOff, Bb + (size_t)k0 * N + n0, 64u, 32u,
                      (unsigned long)N, (unsigned)N, (unsigned)(K - k0));
      __builtin_amdgcn_s_wait_tensorcnt(0);
    }
    __syncthreads();                       // ldsRaw populated
    {
      int kk = tid >> 2;                   // 0..31
      int nb = (tid & 3) * 16;             // 0,16,32,48
      v8h r0 = *(const v8h*)&ldsRaw[kk][nb];
      v8h r1 = *(const v8h*)&ldsRaw[kk][nb + 8];
#pragma unroll
      for (int j = 0; j < 8; ++j) {
        ldsBt[nb + j][kk]     = r0[j];
        ldsBt[nb + 8 + j][kk] = r1[j];
      }
    }
    __syncthreads();                       // ldsBt ready

    v16h a = load_a_frag(A, m0 + (lane & 15), K, k0, lane);
    int kh = (lane >> 4) << 4;
#pragma unroll
    for (int nt = 0; nt < 4; ++nt) {
      const _Float16* bp = &ldsBt[nt * 16 + (lane & 15)][kh >> 1];
      v16h b = pack16(*(const v8h*)bp, *(const v8h*)(bp + 16));
      acc[nt] = __builtin_amdgcn_wmma_f32_16x16x32_f16(false, a, false, b,
                                                       (short)0, acc[nt], false, false);
    }
  }

#pragma unroll
  for (int nt = 0; nt < 4; ++nt) {
#pragma unroll
    for (int r = 0; r < 8; ++r) {
      int m = m0 + r + ((lane >> 4) << 3);
      int n = n0 + nt * 16 + (lane & 15);
      size_t idx = (size_t)m * N + n;
      float val = acc[nt][r];
      if (ACCUM) val += (float)Cb[idx];
      Cb[idx] = (OutT)val;
    }
  }
}

// ---------------------------------------------------------------------------
// Flash attention: one wave per (b, head, 16-query tile).
// qT [B*8][1024][32], kT [B*8][4096][32], vF [B][256][4096] -> attnF [B][256][1024]
// ---------------------------------------------------------------------------
__global__ void attention_wmma(const _Float16* __restrict__ qT,
                               const _Float16* __restrict__ kT,
                               const _Float16* __restrict__ vF,
                               _Float16* __restrict__ attnF,
                               float scale) {
  __shared__ _Float16 ldsP[4][16 * 32];
  int tid = threadIdx.x, lane = tid & 31, wslot = tid >> 5;
  int gw = blockIdx.x * 4 + wslot;        // 0..2047
  int qt = gw & 63;
  int bh = gw >> 6;                       // b*8 + h
  int b = bh >> 3, h = bh & 7;
  int q0 = qt * 16;
  int khalf = (lane >> 4) << 3;

  const _Float16* qbase = qT + ((size_t)bh * NQ + q0 + (lane & 15)) * 32;
  v16h qa = pack16(*(const v8h*)(qbase + 2 * khalf), *(const v8h*)(qbase + 8 + 2 * khalf));

  const _Float16* kbase = kT + (size_t)bh * NK * 32;
  const _Float16* vbase = vF + ((size_t)b * 256 + h * 32) * (size_t)NK;
  _Float16* lp = ldsP[wslot];

  float mrow[8], srow[8];
#pragma unroll
  for (int r = 0; r < 8; ++r) { mrow[r] = -1e30f; srow[r] = 0.0f; }
  v8f oA = {}, oB = {};

  for (int kc = 0; kc < NK; kc += 32) {
    v16h bk0 = *(const v16h*)(kbase + (size_t)(kc + (lane & 15)) * 32 + ((lane >> 4) << 4));
    v16h bk1 = *(const v16h*)(kbase + (size_t)(kc + 16 + (lane & 15)) * 32 + ((lane >> 4) << 4));
    v8f z = {};
    v8f s0 = __builtin_amdgcn_wmma_f32_16x16x32_f16(false, qa, false, bk0, (short)0, z, false, false);
    v8f s1 = __builtin_amdgcn_wmma_f32_16x16x32_f16(false, qa, false, bk1, (short)0, z, false, false);

    // online softmax per row (row q = r + 8*(lane>=16); keys across 16 lanes)
#pragma unroll
    for (int r = 0; r < 8; ++r) {
      float a0 = s0[r] * scale, a1 = s1[r] * scale;
      float t = red16_max(fmaxf(a0, a1));
      float mnew = fmaxf(mrow[r], t);
      float corr = __expf(mrow[r] - mnew);
      float p0 = __expf(a0 - mnew), p1 = __expf(a1 - mnew);
      float u = red16_sum(p0 + p1);
      srow[r] = srow[r] * corr + u;
      mrow[r] = mnew;
      oA[r] *= corr;
      oB[r] *= corr;
      int q = r + ((lane >> 4) << 3);
      lp[q * 32 + (lane & 15)] = (_Float16)p0;
      lp[q * 32 + 16 + (lane & 15)] = (_Float16)p1;
    }
    // wave-private LDS round trip: LDS is in-order per wave, so a dscnt
    // drain is sufficient (no block barrier; waves stay decoupled).
    asm volatile("s_wait_dscnt 0x0" ::: "memory");
    const _Float16* pp = lp + (lane & 15) * 32 + 2 * khalf;
    v16h pa = pack16(*(const v8h*)pp, *(const v8h*)(pp + 8));
    v16h bv0 = *(const v16h*)(vbase + (size_t)(lane & 15) * NK + kc + ((lane >> 4) << 4));
    v16h bv1 = *(const v16h*)(vbase + (size_t)(16 + (lane & 15)) * NK + kc + ((lane >> 4) << 4));
    oA = __builtin_amdgcn_wmma_f32_16x16x32_f16(false, pa, false, bv0, (short)0, oA, false, false);
    oB = __builtin_amdgcn_wmma_f32_16x16x32_f16(false, pa, false, bv1, (short)0, oB, false, false);
  }

#pragma unroll
  for (int r = 0; r < 8; ++r) {
    float inv = 1.0f / srow[r];
    int q = q0 + r + ((lane >> 4) << 3);
    int d = lane & 15;
    attnF[((size_t)b * 256 + h * 32 + d) * NQ + q] = (_Float16)(oA[r] * inv);
    attnF[((size_t)b * 256 + h * 32 + 16 + d) * NQ + q] = (_Float16)(oB[r] * inv);
  }
}

// ---------------------------------------------------------------------------
// Host launcher
// ---------------------------------------------------------------------------
extern "C" void kernel_launch(void* const* d_in, const int* in_sizes, int n_in,
                              void* d_out, int out_size, void* d_ws, size_t ws_size,
                              hipStream_t stream) {
  const float* x  = (const float*)d_in[0];
  const float* g  = (const float*)d_in[1];
  const float* be = (const float*)d_in[2];
  const float* mu = (const float*)d_in[3];
  const float* vr = (const float*)d_in[4];
  const float* w_short = (const float*)d_in[5];
  const float* wq = (const float*)d_in[6];
  const float* wk = (const float*)d_in[7];
  const float* wv = (const float*)d_in[8];
  const float* wo = (const float*)d_in[9];
  float* out = (float*)d_out;

  char* base = (char*)d_ws;
  size_t off = 0;
  auto alloc = [&](size_t halfs) {
    size_t r = off;
    off = (off + halfs * 2 + 255) & ~(size_t)255;
    return (_Float16*)(base + r);
  };
  _Float16* hF    = alloc((size_t)BATCH * CIN * HH * WW);
  _Float16* hsF   = alloc((size_t)BATCH * CIN * NQ);
  _Float16* colF  = alloc((size_t)BATCH * 2304 * NQ);
  _Float16* wsF   = alloc((size_t)COUT * 2304);
  _Float16* wqF   = alloc((size_t)256 * 256);
  _Float16* wkF   = alloc((size_t)256 * 256);
  _Float16* wvF   = alloc((size_t)256 * 256);
  _Float16* woF   = alloc((size_t)COUT * 256);
  _Float16* qF    = alloc((size_t)BATCH * 256 * NQ);
  _Float16* kF    = alloc((size_t)BATCH * 256 * NK);
  _Float16* vF    = alloc((size_t)BATCH * 256 * NK);
  _Float16* qT    = alloc((size_t)BATCH * HEADS * NQ * DK);
  _Float16* kT    = alloc((size_t)BATCH * HEADS * NK * DK);
  _Float16* attnF = alloc((size_t)BATCH * 256 * NQ);

  bn_relu_half<<<(BATCH * CIN * HH * WW) / 256, 256, 0, stream>>>(x, g, be, mu, vr, hF);
  sample_s2<<<(BATCH * CIN * NQ) / 256, 256, 0, stream>>>(hF, hsF);
  cvt_half<<<(COUT * 2304 + 255) / 256, 256, 0, stream>>>(w_short, wsF, COUT * 2304);
  cvt_half<<<(256 * 256 + 255) / 256, 256, 0, stream>>>(wq, wqF, 256 * 256);
  cvt_half<<<(256 * 256 + 255) / 256, 256, 0, stream>>>(wk, wkF, 256 * 256);
  cvt_half<<<(256 * 256 + 255) / 256, 256, 0, stream>>>(wv, wvF, 256 * 256);
  cvt_half<<<(COUT * 256 + 255) / 256, 256, 0, stream>>>(wo, woF, COUT * 256);
  im2col3x3s2<<<(BATCH * 2304 * NQ) / 256, 256, 0, stream>>>(hF, colF);

  dim3 blk(128, 1, 1);
  // K = Wk @ h  [256 x 4096] per batch (fp16 out)
  gemm_wmma<_Float16, false><<<dim3(NK / 64, 256 / 64, BATCH), blk, 0, stream>>>(
      wkF, hF, kF, 256, NK, 256, (long)256 * NK, (long)256 * NK);
  // V = Wv @ h
  gemm_wmma<_Float16, false><<<dim3(NK / 64, 256 / 64, BATCH), blk, 0, stream>>>(
      wvF, hF, vF, 256, NK, 256, (long)256 * NK, (long)256 * NK);
  // Q = Wq @ h_s [256 x 1024] per batch
  gemm_wmma<_Float16, false><<<dim3(NQ / 64, 256 / 64, BATCH), blk, 0, stream>>>(
      wqF, hsF, qF, 256, NQ, 256, (long)256 * NQ, (long)256 * NQ);
  // head transposes for fragment-friendly layouts
  transpose_heads<<<(BATCH * 256 * NQ) / 256, 256, 0, stream>>>(qF, qT, NQ);
  transpose_heads<<<(BATCH * 256 * NK) / 256, 256, 0, stream>>>(kF, kT, NK);
  // flash attention (2048 waves)
  attention_wmma<<<512, blk, 0, stream>>>(qT, kT, vF, attnF, 0.17677669529663687f);
  // shortcut: out = Ws @ im2col [512 x 1024] per batch (f32, overwrite)
  gemm_wmma<float, false><<<dim3(NQ / 64, COUT / 64, BATCH), blk, 0, stream>>>(
      wsF, colF, out, COUT, NQ, 2304, (long)2304 * NQ, (long)COUT * NQ);
  // out += Wo @ attn
  gemm_wmma<float, true><<<dim3(NQ / 64, COUT / 64, BATCH), blk, 0, stream>>>(
      woF, attnF, out, COUT, NQ, 256, (long)256 * NQ, (long)COUT * NQ);

  (void)in_sizes; (void)n_in; (void)out_size; (void)ws_size;
}
